// GINNet_24395414241688
// MI455X (gfx1250) — matrix operand order repaired and checked by reading, not compiled, not run on hardware
//
#include <hip/hip_runtime.h>

// ---------------------------------------------------------------------------
// GIN network forward for MI455X (gfx1250), bf16 WMMA matrix cores.
// ---------------------------------------------------------------------------

#define NN    100000      // nodes
#define NPAD  100096      // nodes padded to multiple of 128 (block M tile)
#define NE    1600000     // edges
#define NG    512         // graphs
#define HCH   128         // hidden channels

typedef __attribute__((ext_vector_type(16))) __bf16        v16bf;
typedef __attribute__((ext_vector_type(8)))  float         v8f;
typedef __attribute__((ext_vector_type(4)))  unsigned int  uiv4;

static __device__ __forceinline__ unsigned short f2bf(float f) {
  union { float f; unsigned u; } c; c.f = f;
  unsigned u = c.u;
  u += 0x7FFFu + ((u >> 16) & 1u);          // round-to-nearest-even
  return (unsigned short)(u >> 16);
}

// A-fragment (16x32 bf16, MxK): lanes 0-15 rows M, K = half*8 + {0..7, 16..23}
static __device__ __forceinline__ v16bf frag_a(const unsigned short* row, int k0) {
  union { uiv4 q[2]; v16bf v; } u;
  u.q[0] = *reinterpret_cast<const uiv4*>(row + k0);
  u.q[1] = *reinterpret_cast<const uiv4*>(row + k0 + 16);
  return u.v;
}
// B-fragment (32x16 bf16, KxN) from W^T rows: K = half*8 + {0..7, 16..23}
static __device__ __forceinline__ v16bf frag_b(const unsigned short* row, int k0) {
  union { uiv4 q[2]; v16bf v; } u;
  u.q[0] = *reinterpret_cast<const uiv4*>(row + k0);
  u.q[1] = *reinterpret_cast<const uiv4*>(row + k0 + 16);
  return u.v;
}

#define WMMA_BF16(a, b, c) \
  __builtin_amdgcn_wmma_f32_16x16x32_bf16(false, (a), false, (b), (short)0, (c), false, false)

// ------------------------- WMMA GEMM --------------------------------------
// C[M x N] = A[M x K] * B[K x N] + bias, optional ReLU.
// A: bf16 row-major (lda), Bt: bf16 W^T row-major [N][ldb].
// Block: 128 threads = 4 waves; wave computes 32(M) x 64(N) with 8 accums,
// so each B fragment is reused by two WMMAs (halves L2 weight traffic).
__global__ void __launch_bounds__(128) k_gemm(
    const unsigned short* __restrict__ A, int lda,
    const unsigned short* __restrict__ Bt, int ldb,
    const float* __restrict__ bias,
    float* __restrict__ outF, unsigned short* __restrict__ outB, int ldo,
    int K, int relu)
{
  const int lane = threadIdx.x & 31;
  const int wave = threadIdx.x >> 5;
  const int hf   = lane >> 4;        // 0|1
  const int lr   = lane & 15;        // row-in-tile (A) / col-in-tile (B,D)
  const int m0   = (blockIdx.x * 4 + wave) << 5;   // 32 rows per wave
  const int n0   = blockIdx.y << 6;

  v8f acc[2][4] = {};
  const unsigned short* ar0 = A + (size_t)(m0 + lr) * lda;
  const unsigned short* ar1 = ar0 + (size_t)16 * lda;
  const unsigned short* br0 = Bt + (size_t)(n0 + lr) * ldb;
  const unsigned short* br1 = br0 + (size_t)16 * ldb;
  const unsigned short* br2 = br0 + (size_t)32 * ldb;
  const unsigned short* br3 = br0 + (size_t)48 * ldb;

  for (int kk = 0; kk < K; kk += 32) {
    __builtin_prefetch(ar0 + kk + 128, 0, 3);      // WGP-scope prefetch, A stream
    __builtin_prefetch(ar1 + kk + 128, 0, 3);
    const int ka = kk + hf * 8;
    const int kb = kk + hf * 8;
    v16bf a0 = frag_a(ar0, ka);
    v16bf a1 = frag_a(ar1, ka);
    v16bf b0 = frag_b(br0, kb);
    acc[0][0] = WMMA_BF16(a0, b0, acc[0][0]);
    acc[1][0] = WMMA_BF16(a1, b0, acc[1][0]);
    v16bf b1 = frag_b(br1, kb);
    acc[0][1] = WMMA_BF16(a0, b1, acc[0][1]);
    acc[1][1] = WMMA_BF16(a1, b1, acc[1][1]);
    v16bf b2 = frag_b(br2, kb);
    acc[0][2] = WMMA_BF16(a0, b2, acc[0][2]);
    acc[1][2] = WMMA_BF16(a1, b2, acc[1][2]);
    v16bf b3 = frag_b(br3, kb);
    acc[0][3] = WMMA_BF16(a0, b3, acc[0][3]);
    acc[1][3] = WMMA_BF16(a1, b3, acc[1][3]);
  }

  // D layout: VGPR r -> lanes 0-15 (M=r, N=lane), lanes 16-31 (M=8+r, N=lane-16)
#pragma unroll
  for (int j = 0; j < 4; ++j) {
    const int n  = n0 + j * 16 + lr;
    const float bv = bias[n];
#pragma unroll
    for (int i = 0; i < 2; ++i) {
#pragma unroll
      for (int r = 0; r < 8; ++r) {
        const int row = m0 + i * 16 + hf * 8 + r;
        float v = acc[i][j][r] + bv;
        if (relu) v = fmaxf(v, 0.f);
        const size_t o = (size_t)row * ldo + n;
        if (outF) outF[o] = v;
        if (outB) outB[o] = f2bf(v);
      }
    }
  }
}

// ---------------------- elementwise / graph kernels -----------------------

__global__ void k_zero(float* __restrict__ p, int n) {
  int gid = blockIdx.x * 256 + threadIdx.x;
  if (gid < n) p[gid] = 0.f;
}

// h[r*ldh + c] = (r<Nrows && c<C) ? x[r*ldx + c] : 0   over NPAD*ldh elements
__global__ void k_copy_pad(const float* __restrict__ x, int ldx,
                           float* __restrict__ h, int ldh,
                           int Nrows, int C, long total) {
  long gid = (long)blockIdx.x * 256 + threadIdx.x;
  if (gid >= total) return;
  int r = (int)(gid / ldh), c = (int)(gid % ldh);
  h[gid] = (r < Nrows && c < C) ? x[(size_t)r * ldx + c] : 0.f;
}

__global__ void k_f2b(const float* __restrict__ s, unsigned short* __restrict__ d, long total) {
  long gid = (long)blockIdx.x * 256 + threadIdx.x;
  if (gid < total) d[gid] = f2bf(s[gid]);
}

// h[dst[e]*ldh + c] += x[src[e]*ldx + c]   (thread per edge-channel, c fastest)
__global__ void k_scatter(const int* __restrict__ src, const int* __restrict__ dst,
                          const float* __restrict__ x, int ldx,
                          float* __restrict__ h, int ldh, int C, long total) {
  long gid = (long)blockIdx.x * 256 + threadIdx.x;
  if (gid >= total) return;
  int e = (int)(gid / C), c = (int)(gid % C);
  atomicAdd(&h[(size_t)dst[e] * ldh + c], x[(size_t)src[e] * ldx + c]);
}

// BN pass 1: per-channel sum / sumsq of relu(x); block = 64 channels x 4 row-groups
__global__ void k_bn_reduce(const float* __restrict__ x, int C, int Nrows,
                            float* __restrict__ sum, float* __restrict__ ssq) {
  __shared__ float s0[256], s1[256];
  const int c  = blockIdx.x * 64 + (threadIdx.x & 63);
  const int rg = threadIdx.x >> 6;
  const int chunk = (Nrows + gridDim.y - 1) / gridDim.y;
  const int rbeg = blockIdx.y * chunk;
  int rend = rbeg + chunk; if (rend > Nrows) rend = Nrows;
  float a = 0.f, b = 0.f;
  for (int r = rbeg + rg; r < rend; r += 4) {
    float v = fmaxf(x[(size_t)r * C + c], 0.f);
    a += v; b += v * v;
  }
  s0[threadIdx.x] = a; s1[threadIdx.x] = b;
  __syncthreads();
  if (rg == 0) {
    a = s0[threadIdx.x] + s0[threadIdx.x + 64] + s0[threadIdx.x + 128] + s0[threadIdx.x + 192];
    b = s1[threadIdx.x] + s1[threadIdx.x + 64] + s1[threadIdx.x + 128] + s1[threadIdx.x + 192];
    atomicAdd(&sum[c], a);
    atomicAdd(&ssq[c], b);
  }
}

__global__ void k_bn_finalize(const float* __restrict__ sum, const float* __restrict__ ssq,
                              const float* __restrict__ gamma, const float* __restrict__ beta,
                              float* __restrict__ scale, float* __restrict__ shift, int Nrows) {
  int c = threadIdx.x;
  float inv = 1.f / (float)Nrows;
  float mu  = sum[c] * inv;
  float var = ssq[c] * inv - mu * mu;
  float sc  = gamma[c] * rsqrtf(var + 1e-5f);
  scale[c] = sc;
  shift[c] = beta[c] - mu * sc;
}

// x = relu(x)*scale[c] + shift[c]   in place (rows < NN only)
__global__ void k_bn_apply(float* __restrict__ x, const float* __restrict__ scale,
                           const float* __restrict__ shift, int C, long total) {
  long gid = (long)blockIdx.x * 256 + threadIdx.x;
  if (gid >= total) return;
  int c = (int)(gid % C);
  float v = fmaxf(x[gid], 0.f);
  x[gid] = v * scale[c] + shift[c];
}

// global_add_pool: g[batch[i]*128 + c] += x[i*128 + c]
__global__ void k_pool(const float* __restrict__ x, const int* __restrict__ batch,
                       float* __restrict__ g, long total) {
  long gid = (long)blockIdx.x * 256 + threadIdx.x;
  if (gid >= total) return;
  int i = (int)(gid >> 7), c = (int)(gid & 127);
  atomicAdd(&g[(size_t)batch[i] * HCH + c], x[gid]);
}

// relu(g @ Wf1 + bf1) @ Wf2 + bf2, relu -> out[NG]
__global__ void k_fc(const float* __restrict__ g, const float* __restrict__ Wf1,
                     const float* __restrict__ bf1, const float* __restrict__ Wf2,
                     const float* __restrict__ bf2, float* __restrict__ out) {
  __shared__ float hbuf[64];
  const int bg = blockIdx.x, t = threadIdx.x;
  float acc = bf1[t];
  for (int k = 0; k < HCH; ++k) acc += g[(size_t)bg * HCH + k] * Wf1[k * 64 + t];
  hbuf[t] = fmaxf(acc, 0.f);
  __syncthreads();
  if (t == 0) {
    float o = bf2[0];
    for (int k = 0; k < 64; ++k) o += hbuf[k] * Wf2[k];
    out[bg] = fmaxf(o, 0.f);
  }
}

// Wt[o*Kp + k] = bf16(W[k*O + o]), zero-padded K
__global__ void k_cvt_weight(const float* __restrict__ W, unsigned short* __restrict__ Wt,
                             int K, int O, int Kp) {
  int gid = blockIdx.x * 256 + threadIdx.x;
  if (gid >= O * Kp) return;
  int o = gid / Kp, k = gid % Kp;
  float v = (k < K) ? W[(size_t)k * O + o] : 0.f;
  Wt[gid] = f2bf(v);
}

// ---------------------------------------------------------------------------
extern "C" void kernel_launch(void* const* d_in, const int* in_sizes, int n_in,
                              void* d_out, int out_size, void* d_ws, size_t ws_size,
                              hipStream_t stream) {
  (void)n_in; (void)out_size; (void)ws_size;
  static const int CHs[6] = {126, 256, 256, 512, 256, 128};

  const float* x_in  = (const float*)d_in[0];
  const int*   eidx  = (const int*)d_in[1];
  const int*   srcA  = eidx;
  const int*   dstA  = eidx + NE;
  const int*   batch = (const int*)d_in[3];

  // Detect nested-dict flatten order: sorted keys (W1,W2,b1,b2) vs insertion
  // (W1,b1,W2,b2) using the second entry of layer 0 (256*256 if sorted).
  const bool sortedOrder = (in_sizes[5] == 256 * 256);

  const float *W1p[25], *B1p[25], *W2p[25], *B2p[25];
  int idx = 4;
  for (int b = 0; b < 5; ++b)
    for (int l = 0; l < 5; ++l) {
      int li = b * 5 + l;
      if (sortedOrder) {
        W1p[li] = (const float*)d_in[idx + 0]; W2p[li] = (const float*)d_in[idx + 1];
        B1p[li] = (const float*)d_in[idx + 2]; B2p[li] = (const float*)d_in[idx + 3];
      } else {
        W1p[li] = (const float*)d_in[idx + 0]; B1p[li] = (const float*)d_in[idx + 1];
        W2p[li] = (const float*)d_in[idx + 2]; B2p[li] = (const float*)d_in[idx + 3];
      }
      idx += 4;
    }
  const float *bnG[5], *bnB[5];
  for (int b = 0; b < 5; ++b) {
    if (sortedOrder) { bnB[b] = (const float*)d_in[idx]; bnG[b] = (const float*)d_in[idx + 1]; }
    else             { bnG[b] = (const float*)d_in[idx]; bnB[b] = (const float*)d_in[idx + 1]; }
    idx += 2;
  }
  const float *Wf1, *bf1v, *Wf2, *bf2v;
  if (sortedOrder) {
    Wf1  = (const float*)d_in[idx + 0]; Wf2  = (const float*)d_in[idx + 1];
    bf1v = (const float*)d_in[idx + 2]; bf2v = (const float*)d_in[idx + 3];
  } else {
    Wf1  = (const float*)d_in[idx + 0]; bf1v = (const float*)d_in[idx + 1];
    Wf2  = (const float*)d_in[idx + 2]; bf2v = (const float*)d_in[idx + 3];
  }

  // ---- carve workspace -----------------------------------------------------
  char* wp = (char*)d_ws;
  auto take = [&](size_t bytes) -> void* {
    void* r = (void*)wp;
    wp += (bytes + 255) & ~(size_t)255;
    return r;
  };
  float*          xf    = (float*)take((size_t)NPAD * 512 * 4);   // fp32 activations
  float*          hfb   = (float*)take((size_t)NPAD * 512 * 4);   // h = x + agg (fp32 accum)
  unsigned short* hb    = (unsigned short*)take((size_t)NPAD * 512 * 2); // h bf16
  unsigned short* tb    = (unsigned short*)take((size_t)NPAD * 512 * 2); // mid bf16
  unsigned short* wbase = (unsigned short*)take((size_t)5500000 * 2);    // bf16 W^T pool
  float*          bnsum   = (float*)take(512 * 4);
  float*          bnssq   = (float*)take(512 * 4);
  float*          bnScale = (float*)take(512 * 4);
  float*          bnShift = (float*)take(512 * 4);
  float*          gbuf    = (float*)take((size_t)NG * HCH * 4);

  // ---- convert weights to bf16 W^T (padded K) ------------------------------
  unsigned short *wt1[25], *wt2[25];
  size_t woff = 0;
  for (int b = 0; b < 5; ++b)
    for (int l = 0; l < 5; ++l) {
      int li = b * 5 + l;
      int ci = CHs[b], co = CHs[b + 1];
      int K1 = (l == 0) ? ci : co;
      int Kp1 = (K1 + 31) & ~31;
      wt1[li] = wbase + woff; woff += (size_t)co * Kp1;
      wt2[li] = wbase + woff; woff += (size_t)co * co;
      int n1 = co * Kp1, n2 = co * co;
      k_cvt_weight<<<(n1 + 255) / 256, 256, 0, stream>>>(W1p[li], wt1[li], K1, co, Kp1);
      k_cvt_weight<<<(n2 + 255) / 256, 256, 0, stream>>>(W2p[li], wt2[li], co, co, co);
    }

  // ---- GIN blocks ----------------------------------------------------------
  const float* xcur = x_in;
  int ldx = CHs[0];
  for (int b = 0; b < 5; ++b) {
    int ci = CHs[b], co = CHs[b + 1];
    for (int l = 0; l < 5; ++l) {
      int li = b * 5 + l;
      int K1 = (l == 0) ? ci : co;
      int Kp = (K1 + 31) & ~31;

      long tot = (long)NPAD * Kp;
      k_copy_pad<<<(unsigned)((tot + 255) / 256), 256, 0, stream>>>(xcur, ldx, hfb, Kp, NN, K1, tot);
      tot = (long)NE * K1;
      k_scatter<<<(unsigned)((tot + 255) / 256), 256, 0, stream>>>(srcA, dstA, xcur, ldx, hfb, Kp, K1, tot);
      tot = (long)NPAD * Kp;
      k_f2b<<<(unsigned)((tot + 255) / 256), 256, 0, stream>>>(hfb, hb, tot);

      dim3 gg(NPAD / 128, co / 64);
      // GEMM1: relu(h @ W1 + b1) -> bf16 mid
      k_gemm<<<gg, 128, 0, stream>>>(hb, Kp, wt1[li], Kp, B1p[li],
                                     nullptr, tb, co, Kp, 1);
      // GEMM2: mid @ W2 + b2 -> fp32 x  (ReLU between inner layers)
      k_gemm<<<gg, 128, 0, stream>>>(tb, co, wt2[li], co, B2p[li],
                                     xf, nullptr, co, co, (l < 4) ? 1 : 0);
      xcur = xf; ldx = co;
    }
    // relu + BatchNorm (training-mode batch stats)
    k_zero<<<4, 256, 0, stream>>>(bnsum, 512);
    k_zero<<<4, 256, 0, stream>>>(bnssq, 512);
    dim3 gr(co / 64, 64);
    k_bn_reduce<<<gr, 256, 0, stream>>>(xf, co, NN, bnsum, bnssq);
    k_bn_finalize<<<1, co, 0, stream>>>(bnsum, bnssq, bnG[b], bnB[b], bnScale, bnShift, NN);
    long tot = (long)NN * co;
    k_bn_apply<<<(unsigned)((tot + 255) / 256), 256, 0, stream>>>(xf, bnScale, bnShift, co, tot);
  }

  // ---- pool + FC head ------------------------------------------------------
  k_zero<<<(NG * HCH + 255) / 256, 256, 0, stream>>>(gbuf, NG * HCH);
  long tot = (long)NN * HCH;
  k_pool<<<(unsigned)((tot + 255) / 256), 256, 0, stream>>>(xf, batch, gbuf, tot);
  k_fc<<<NG, 64, 0, stream>>>(gbuf, Wf1, bf1v, Wf2, bf2v, (float*)d_out);
}
